// CGCNNModel_55224689492319
// MI455X (gfx1250) — compile-verified, matrix operand order checked by reading, not run on hardware
//
#include <hip/hip_runtime.h>
#include <hip/hip_bf16.h>

// ---------------- problem constants ----------------
#define N_NODES   50000
#define N_EDGES   1600000
#define NUM_GRAPHS 512
#define HID       128
#define EDGE_DIM  64
#define N_CONV    4
#define ZOUT      512          // [f_dst | f_src | g_dst | g_src]
#define BN_EPS    1e-5f

#define EDGE_GROUPS   (N_EDGES / 16)   // 100000
#define EDGE_BLOCKS   1250             // x 8 waves = 10000 waves, grid-stride
#define EDGE_WAVES    (EDGE_BLOCKS * 8)

typedef __attribute__((ext_vector_type(16))) __bf16 v16bf;
typedef __attribute__((ext_vector_type(8)))  float  v8f;

// ---- WMMA bf16 fragment loaders (per CDNA5 ISA §7.12.2 layouts) ----
// 16-bit A (16x32, MxK): lanes 0-15 and 16-31 both hold rows M=lane&15.
// half=lane>>4 selects K chunk: VGPR0-3 = K[half*8 .. half*8+7],
// VGPR4-7 = K[16+half*8 .. 16+half*8+7].
// B (32x16, KxN) loaded from an N-major (transposed) weight array uses the
// identical pattern with "row" = output column. Works for global or LDS ptrs.
__device__ __forceinline__ v16bf load_frag_bf16(const __bf16* base,
                                                int ld, int k0, int lane) {
    const int r    = lane & 15;
    const int half = lane >> 4;
    const __bf16* p = base + (long long)r * ld + k0 + half * 8;
    union { uint4 q[2]; v16bf v; } u;
    u.q[0] = *(const uint4*)(p);
    u.q[1] = *(const uint4*)(p + 16);
    return u.v;
}

// A fragment for the edge GEMM: rows are edges, fp32 edge_attr converted to
// bf16 in registers (RNE via compiler fptrunc -> v_cvt bf16).
__device__ __forceinline__ v16bf load_edgeA(const float* __restrict__ eattr,
                                            long long e0, int k0, int lane) {
    const int m    = lane & 15;
    const int half = lane >> 4;
    const float* p = eattr + (e0 + m) * EDGE_DIM + k0 + half * 8;
    float4 f0 = *(const float4*)(p);
    float4 f1 = *(const float4*)(p + 4);
    float4 f2 = *(const float4*)(p + 16);
    float4 f3 = *(const float4*)(p + 20);
    v16bf v;
    v[0]=(__bf16)f0.x;  v[1]=(__bf16)f0.y;  v[2]=(__bf16)f0.z;  v[3]=(__bf16)f0.w;
    v[4]=(__bf16)f1.x;  v[5]=(__bf16)f1.y;  v[6]=(__bf16)f1.z;  v[7]=(__bf16)f1.w;
    v[8]=(__bf16)f2.x;  v[9]=(__bf16)f2.y;  v[10]=(__bf16)f2.z; v[11]=(__bf16)f2.w;
    v[12]=(__bf16)f3.x; v[13]=(__bf16)f3.y; v[14]=(__bf16)f3.z; v[15]=(__bf16)f3.w;
    return v;
}

__device__ __forceinline__ v8f wmma_bf16(v16bf a, v16bf b, v8f c) {
    return __builtin_amdgcn_wmma_f32_16x16x32_bf16(false, a, false, b,
                                                   (short)0, c, false, false);
}

// ---------------- weight prep (fp32 -> transposed bf16) ----------------
// W4t[l][o][k], o in [0,512): o<128 -> Wf[l][k][o]       (f, dst)
//                             o<256 -> Wf[l][128+k][o-128](f, src)
//                             o<384 -> Ws[l][k][o-256]    (s, dst)
//                             else  -> Ws[l][128+k][o-384](s, src)
__global__ void prep_w4_kernel(const float* __restrict__ Wf,
                               const float* __restrict__ Ws,
                               __bf16* __restrict__ W4t) {
    int idx = blockIdx.x * blockDim.x + threadIdx.x;   // 4*512*128
    int l = idx >> 16;              // /65536
    int rem = idx & 65535;
    int o = rem >> 7;
    int k = rem & 127;
    const float* base = (o < 256 ? Wf : Ws) + (long long)l * 320 * 128;
    int oo = o & 127;
    int row = (o & 128) ? (128 + k) : k;
    W4t[idx] = (__bf16)base[row * 128 + oo];
}

// Wet[l][p][n][k] = W{f,s}[l][256+k][n], k in [0,64)
__global__ void prep_we_kernel(const float* __restrict__ Wf,
                               const float* __restrict__ Ws,
                               __bf16* __restrict__ Wet) {
    int idx = blockIdx.x * blockDim.x + threadIdx.x;   // 4*2*128*64
    int l = idx >> 14;              // /16384
    int rem = idx & 16383;
    int p = rem >> 13;
    int rem2 = rem & 8191;
    int n = rem2 >> 6;
    int k = rem2 & 63;
    const float* base = (p ? Ws : Wf) + (long long)l * 320 * 128;
    Wet[idx] = (__bf16)base[(256 + k) * 128 + n];
}

// ---------------- node embed: h = x @ W_emb + b_emb ----------------
__global__ void embed_kernel(const float* __restrict__ x,
                             const float* __restrict__ Wemb,
                             const float* __restrict__ bemb,
                             float* __restrict__ h, __bf16* __restrict__ hbf) {
    int idx = blockIdx.x * blockDim.x + threadIdx.x;   // N*128
    int n = idx >> 7, c = idx & 127;
    float acc = bemb[c];
    const float* xr = x + n * 9;
#pragma unroll
    for (int k = 0; k < 9; ++k) acc += xr[k] * Wemb[k * 128 + c];
    h[idx] = acc;
    hbf[idx] = (__bf16)acc;
}

// ---------------- degree / graph-count ----------------
__global__ void deg_count_kernel(const long long* __restrict__ dst,
                                 float* __restrict__ deg) {
    int e = blockIdx.x * blockDim.x + threadIdx.x;     // N_EDGES
    atomicAdd(&deg[(int)dst[e]], 1.0f);
}
__global__ void deg_inv_kernel(float* __restrict__ deg) {
    int n = blockIdx.x * blockDim.x + threadIdx.x;
    if (n < N_NODES) deg[n] = 1.0f / fmaxf(deg[n], 1.0f);
}
__global__ void gcnt_kernel(const long long* __restrict__ batch,
                            float* __restrict__ gcnt) {
    int n = blockIdx.x * blockDim.x + threadIdx.x;
    if (n < N_NODES) atomicAdd(&gcnt[(int)batch[n]], 1.0f);
}

// ---------------- node projection: P[N,512] = h_bf16 @ W4t^T ----------------
// One wave computes a 16x64 output strip (4 N-tiles): the A fragment is loaded
// once per K-step and reused by 4 WMMAs -> 16 WMMAs/wave, 4x less A traffic.
// 3125 M-tiles x 8 strips = 25000 waves (8 waves/block -> 3125 blocks, exact).
__global__ void node_proj_kernel(const __bf16* __restrict__ hbf,
                                 const __bf16* __restrict__ W4l,
                                 float* __restrict__ P) {
    int wave = blockIdx.x * (blockDim.x >> 5) + (threadIdx.x >> 5);
    int mt = wave >> 3;            // /8 : node tile
    int q  = wave & 7;             // strip: N-tiles q*4 .. q*4+3
    int lane = threadIdx.x & 31;
    const __bf16* Arow = hbf + (long long)(mt * 16) * HID;

    v8f c0 = {0.f,0.f,0.f,0.f,0.f,0.f,0.f,0.f};
    v8f c1 = c0, c2 = c0, c3 = c0;
#pragma unroll
    for (int kb = 0; kb < 4; ++kb) {
        int k0 = kb * 32;
        v16bf a = load_frag_bf16(Arow, HID, k0, lane);
        const __bf16* Bb = W4l + (long long)(q * 64) * HID;
        c0 = wmma_bf16(a, load_frag_bf16(Bb,            HID, k0, lane), c0);
        c1 = wmma_bf16(a, load_frag_bf16(Bb + 16 * HID, HID, k0, lane), c1);
        c2 = wmma_bf16(a, load_frag_bf16(Bb + 32 * HID, HID, k0, lane), c2);
        c3 = wmma_bf16(a, load_frag_bf16(Bb + 48 * HID, HID, k0, lane), c3);
    }
    int half = lane >> 4, n = lane & 15;
    long long col0 = q * 64 + n;
#pragma unroll
    for (int r = 0; r < 8; ++r) {
        long long row = (long long)(mt * 16 + r + 8 * half) * ZOUT;
        P[row + col0     ] = c0[r];
        P[row + col0 + 16] = c1[r];
        P[row + col0 + 32] = c2[r];
        P[row + col0 + 48] = c3[r];
    }
}

// ---------------- fused edge kernel ----------------
// Layer weights (2x128x64 bf16 = 32KB) are staged in LDS once per block; each
// wave grid-strides over 16-edge groups (~10 groups/wave), doing the edge GEMM
// with WMMA (B fragments from LDS -> ds_load_b128), gathering node projections,
// sigmoid*softplus, and scaled atomic scatter into aggr.
__global__ void edge_msg_kernel(const float* __restrict__ eattr,
                                const long long* __restrict__ src,
                                const long long* __restrict__ dst,
                                const float* __restrict__ P,
                                const __bf16* __restrict__ We_l,   // [2][128][64]
                                const float* __restrict__ biasF,
                                const float* __restrict__ biasS,
                                const float* __restrict__ invdeg,
                                float* __restrict__ aggr) {
    __shared__ __align__(16) __bf16 sWe[2 * HID * EDGE_DIM];   // 32 KB
    // cooperative stage: 2048 uint4 over 256 threads
    {
        const uint4* gsrc = (const uint4*)We_l;
        uint4* ldst = (uint4*)sWe;
        for (int i = threadIdx.x; i < (2 * HID * EDGE_DIM) / 8; i += 256)
            ldst[i] = gsrc[i];
    }
    __syncthreads();

    int wave0 = blockIdx.x * (blockDim.x >> 5) + (threadIdx.x >> 5);
    int lane = threadIdx.x & 31;
    int half = lane >> 4;
    int n = lane & 15;
    const __bf16* sWeF = sWe;                 // p=0
    const __bf16* sWeS = sWe + HID * EDGE_DIM;

    // g-invariant per-lane bias values for the 8 N-tiles
    float bvF[8], bvS[8];
#pragma unroll
    for (int nt = 0; nt < 8; ++nt) {
        bvF[nt] = biasF[nt * 16 + n];
        bvS[nt] = biasS[nt * 16 + n];
    }

    for (int g = wave0; g < EDGE_GROUPS; g += EDGE_WAVES) {
        long long e0 = (long long)g * 16;

        // edge endpoints for the 8 C-rows this lane owns (M = r + 8*half)
        int dr[8], sr[8];
        float iv[8];
#pragma unroll
        for (int r = 0; r < 8; ++r) {
            long long e = e0 + r + 8 * half;
            dr[r] = (int)dst[e];
            sr[r] = (int)src[e];
            iv[r] = invdeg[dr[r]];
        }

        // A fragments (shared by f and g projections), fp32->bf16 in registers
        v16bf a0 = load_edgeA(eattr, e0, 0, lane);
        v16bf a1 = load_edgeA(eattr, e0, 32, lane);

#pragma unroll
        for (int nt = 0; nt < 8; ++nt) {
            int col = nt * 16 + n;
            float bF = bvF[nt], bS = bvS[nt];
            v8f fc = {bF,bF,bF,bF,bF,bF,bF,bF};
            v8f gc = {bS,bS,bS,bS,bS,bS,bS,bS};

            fc = wmma_bf16(a0, load_frag_bf16(sWeF + nt*16*EDGE_DIM, EDGE_DIM, 0,  lane), fc);
            fc = wmma_bf16(a1, load_frag_bf16(sWeF + nt*16*EDGE_DIM, EDGE_DIM, 32, lane), fc);
            gc = wmma_bf16(a0, load_frag_bf16(sWeS + nt*16*EDGE_DIM, EDGE_DIM, 0,  lane), gc);
            gc = wmma_bf16(a1, load_frag_bf16(sWeS + nt*16*EDGE_DIM, EDGE_DIM, 32, lane), gc);

            // gather node projections: 16-lane groups read 16 consecutive
            // columns of one P row -> coalesced 64B segments, L2-resident
#pragma unroll
            for (int r = 0; r < 8; ++r) {
                long long pd = (long long)dr[r] * ZOUT;
                long long ps = (long long)sr[r] * ZOUT;
                float f = fc[r] + P[pd + col]       + P[ps + 128 + col];
                float gg = gc[r] + P[pd + 256 + col] + P[ps + 384 + col];
                float s  = 1.0f / (1.0f + __expf(-f));
                float sp = fmaxf(gg, 0.0f) + log1pf(__expf(-fabsf(gg)));
                float msg = s * sp * iv[r];
                atomicAdd(&aggr[(long long)dr[r] * HID + col], msg);
            }
        }
    }
}

// ---------------- BN stats: h_conv = aggr + h (in place), sum/sumsq ----------
__global__ void bn_stats_kernel(float* __restrict__ aggr,
                                const float* __restrict__ h,
                                float* __restrict__ bnsum) {
    int col = threadIdx.x;                 // 128 threads
    float s = 0.f, q = 0.f;
    for (int row = blockIdx.x; row < N_NODES; row += gridDim.x) {
        long long i = (long long)row * HID + col;
        float v = aggr[i] + h[i];
        aggr[i] = v;
        s += v;
        q += v * v;
    }
    atomicAdd(&bnsum[col], s);
    atomicAdd(&bnsum[HID + col], q);
}

__global__ void bn_final_kernel(const float* __restrict__ bnsum,
                                const float* __restrict__ gamma,
                                const float* __restrict__ beta,
                                float* __restrict__ bnscale) {
    int c = threadIdx.x;
    float invN = 1.0f / (float)N_NODES;
    float mu  = bnsum[c] * invN;
    float var = bnsum[HID + c] * invN - mu * mu;
    float rstd = rsqrtf(var + BN_EPS);
    float scale = gamma[c] * rstd;
    bnscale[c] = scale;
    bnscale[HID + c] = beta[c] - mu * scale;
}

// h_new = relu(h_conv*scale + shift) + h_old ; write fp32 + bf16 copies
__global__ void bn_apply_kernel(const float* __restrict__ aggr,
                                const float* __restrict__ bnscale,
                                float* __restrict__ h,
                                __bf16* __restrict__ hbf) {
    int idx = blockIdx.x * blockDim.x + threadIdx.x;   // N*128
    int c = idx & 127;
    float bn = aggr[idx] * bnscale[c] + bnscale[HID + c];
    float hn = fmaxf(bn, 0.0f) + h[idx];
    h[idx] = hn;
    hbf[idx] = (__bf16)hn;
}

// ---------------- pooling + head MLP ----------------
__global__ void pool_kernel(const float* __restrict__ h,
                            const long long* __restrict__ batch,
                            float* __restrict__ pool) {
    int idx = blockIdx.x * blockDim.x + threadIdx.x;   // N*128
    int n = idx >> 7, c = idx & 127;
    atomicAdd(&pool[(long long)batch[n] * HID + c], h[idx]);
}

__global__ void mlp_kernel(const float* __restrict__ pool,
                           const float* __restrict__ gcnt,
                           const float* __restrict__ W1,
                           const float* __restrict__ b1,
                           const float* __restrict__ W2,
                           const float* __restrict__ b2,
                           float* __restrict__ out) {
    int g = blockIdx.x;          // 512 graphs, one wave each
    int lane = threadIdx.x;      // 32 lanes, 2 hidden units each
    float ic = 1.0f / fmaxf(gcnt[g], 1.0f);
    float h0 = b1[lane], h1 = b1[lane + 32];
    const float* pr = pool + (long long)g * HID;
    for (int k = 0; k < HID; ++k) {
        float pm = pr[k] * ic;
        h0 += pm * W1[k * 64 + lane];
        h1 += pm * W1[k * 64 + lane + 32];
    }
    h0 = fmaxf(h0, 0.f) + log1pf(__expf(-fabsf(h0)));
    h1 = fmaxf(h1, 0.f) + log1pf(__expf(-fabsf(h1)));
    float part = h0 * W2[lane] + h1 * W2[lane + 32];
    for (int off = 16; off > 0; off >>= 1)
        part += __shfl_down(part, off, 32);
    if (lane == 0) out[g] = part + b2[0];
}

// ---------------- host orchestration ----------------
extern "C" void kernel_launch(void* const* d_in, const int* in_sizes, int n_in,
                              void* d_out, int out_size, void* d_ws, size_t ws_size,
                              hipStream_t stream) {
    const float*     x     = (const float*)d_in[0];
    const long long* ei    = (const long long*)d_in[1];
    const float*     eattr = (const float*)d_in[2];
    const long long* batch = (const long long*)d_in[3];
    const float*     Wemb  = (const float*)d_in[4];
    const float*     bemb  = (const float*)d_in[5];
    const float*     Wf    = (const float*)d_in[6];
    const float*     bf    = (const float*)d_in[7];
    const float*     Ws    = (const float*)d_in[8];
    const float*     bs    = (const float*)d_in[9];
    const float*     gamma = (const float*)d_in[10];
    const float*     beta  = (const float*)d_in[11];
    const float*     W1    = (const float*)d_in[12];
    const float*     b1    = (const float*)d_in[13];
    const float*     W2    = (const float*)d_in[14];
    const float*     b2    = (const float*)d_in[15];
    float* out = (float*)d_out;

    const long long* src = ei;
    const long long* dst = ei + N_EDGES;

    // workspace carve-up (all 16B-aligned chunks)
    char* p = (char*)d_ws;
    float*  h      = (float*)p;            p += (size_t)N_NODES * HID * 4;     // 25.6 MB
    __bf16* hbf    = (__bf16*)p;           p += (size_t)N_NODES * HID * 2;     // 12.8 MB
    float*  P      = (float*)p;            p += (size_t)N_NODES * ZOUT * 4;    // 102.4 MB
    float*  aggr   = (float*)p;            p += (size_t)N_NODES * HID * 4;     // 25.6 MB
    float*  deg    = (float*)p;            p += (size_t)N_NODES * 4;           // 0.2 MB
    float*  bnsum  = (float*)p;            p += 256 * 4;
    float*  bnscale= (float*)p;            p += 256 * 4;
    __bf16* W4t    = (__bf16*)p;           p += (size_t)N_CONV * ZOUT * HID * 2;     // 512 KB
    __bf16* Wet    = (__bf16*)p;           p += (size_t)N_CONV * 2 * HID * EDGE_DIM * 2; // 128 KB
    float*  pool   = (float*)p;            p += (size_t)NUM_GRAPHS * HID * 4;  // 256 KB
    float*  gcnt   = (float*)p;            p += (size_t)NUM_GRAPHS * 4;

    // per-call zero init (graph-capture safe stream memsets)
    hipMemsetAsync(deg,  0, (size_t)N_NODES * 4, stream);
    hipMemsetAsync(gcnt, 0, (size_t)NUM_GRAPHS * 4, stream);
    hipMemsetAsync(pool, 0, (size_t)NUM_GRAPHS * HID * 4, stream);

    // weight prep + embed + degree/graph counts
    prep_w4_kernel<<<(N_CONV * ZOUT * HID) / 256, 256, 0, stream>>>(Wf, Ws, W4t);
    prep_we_kernel<<<(N_CONV * 2 * HID * EDGE_DIM) / 256, 256, 0, stream>>>(Wf, Ws, Wet);
    embed_kernel<<<(N_NODES * HID) / 256, 256, 0, stream>>>(x, Wemb, bemb, h, hbf);
    deg_count_kernel<<<N_EDGES / 256, 256, 0, stream>>>(dst, deg);
    deg_inv_kernel<<<(N_NODES + 255) / 256, 256, 0, stream>>>(deg);
    gcnt_kernel<<<(N_NODES + 255) / 256, 256, 0, stream>>>(batch, gcnt);

    for (int l = 0; l < N_CONV; ++l) {
        hipMemsetAsync(aggr,  0, (size_t)N_NODES * HID * 4, stream);
        hipMemsetAsync(bnsum, 0, 256 * 4, stream);

        // P = h @ [Wf_d|Wf_s|Ws_d|Ws_s] : 25000 waves x 16 WMMAs, 8 waves/block
        node_proj_kernel<<<3125, 256, 0, stream>>>(
            hbf, W4t + (size_t)l * ZOUT * HID, P);

        // fused edge GEMM (LDS-staged weights) + gather + act + mean-scatter
        edge_msg_kernel<<<EDGE_BLOCKS, 256, 0, stream>>>(
            eattr, src, dst, P, Wet + (size_t)l * 2 * HID * EDGE_DIM,
            bf + l * HID, bs + l * HID, deg, aggr);

        bn_stats_kernel<<<512, 128, 0, stream>>>(aggr, h, bnsum);
        bn_final_kernel<<<1, 128, 0, stream>>>(bnsum, gamma + l * HID,
                                               beta + l * HID, bnscale);
        bn_apply_kernel<<<(N_NODES * HID) / 256, 256, 0, stream>>>(aggr, bnscale, h, hbf);
    }

    pool_kernel<<<(N_NODES * HID) / 256, 256, 0, stream>>>(h, batch, pool);
    mlp_kernel<<<NUM_GRAPHS, 32, 0, stream>>>(pool, gcnt, W1, b1, W2, b2, out);
}
